// GNNCWT2D_Mk11_1sec_3H_83356725281184
// MI455X (gfx1250) — matrix operand order repaired and checked by reading, not compiled, not run on hardware
//
#include <hip/hip_runtime.h>
#include <hip/hip_bf16.h>

// ---------------------------------------------------------------------------
// GNN pipeline for MI455X (gfx1250): bf16 WMMA GEMMs + async global->LDS
// weight staging (ASYNCcnt), stream-ordered multi-kernel, graph-capture safe.
// ---------------------------------------------------------------------------

typedef __attribute__((ext_vector_type(16))) __bf16 v16bf;
typedef __attribute__((ext_vector_type(8)))  __bf16 v8bf;
typedef __attribute__((ext_vector_type(8)))  float  v8f;

union ABf { v16bf v; v8bf h[2]; };

#define NB 4096          // graphs
#define NEL 19           // electrodes per graph
#define NROWS (NB * NEL) // 77824
#define EPG 60           // edges per graph
#define BN_EPS 1e-5f

// Async 16B global->LDS copy (CDNA5 TDM-adjacent path, ASYNCcnt-tracked).
// lds_off: wave-relative LDS byte offset (low 32 bits of flat shared addr).
__device__ __forceinline__ void async_g2l_b128(unsigned lds_off, unsigned long long gaddr) {
  asm volatile("global_load_async_to_lds_b128 %0, %1, off"
               :: "v"(lds_off), "v"(gaddr) : "memory");
}
__device__ __forceinline__ void wait_async0() {
  asm volatile("s_wait_asynccnt 0x0" ::: "memory");
}

// ------------------------------ workspace map ------------------------------
static constexpr size_t OFF_W2T = 4096;                    // 256x800 bf16
static constexpr size_t OFF_W3T = OFF_W2T + 256*800*2;     // 128x256 bf16
static constexpr size_t OFF_W4T = OFF_W3T + 128*256*2;     // 64x128 bf16
static constexpr size_t OFF_WC1 = OFF_W4T + 64*128*2;      // 32x128 bf16
static constexpr size_t OFF_WC2 = OFF_WC1 + 32*128*2;      // 32x64 bf16
static constexpr size_t OFF_H1  = 524288;                              // [R][256] bf16
static constexpr size_t OFF_H2  = OFF_H1 + (size_t)NROWS*256*2;        // [R][128] bf16
static constexpr size_t OFF_H3  = OFF_H2 + (size_t)NROWS*128*2;        // [R][64]  bf16
static constexpr size_t OFF_A4  = OFF_H3 + (size_t)NROWS*64*2;         // [R][128] bf16
static constexpr size_t OFF_H4  = OFF_A4 + (size_t)NROWS*128*2;        // [R][32]  bf16
static constexpr size_t OFF_A5  = OFF_H4 + (size_t)NROWS*32*2;         // [R][64]  bf16
static constexpr size_t OFF_H5  = OFF_A5 + (size_t)NROWS*64*2;         // [R][32]  bf16

// ------------------------------ small helpers ------------------------------
__global__ void k_zero(float* __restrict__ p, int n) {
  int i = blockIdx.x * blockDim.x + threadIdx.x;
  if (i < n) p[i] = 0.0f;
}

// Wt[n][k] = (bf16) W[k][n]   (W is [FIN][FOUT] row-major)
template <int FIN, int FOUT>
__global__ void k_wt(const float* __restrict__ W, __bf16* __restrict__ Wt) {
  int i = blockIdx.x * blockDim.x + threadIdx.x;
  if (i < FIN * FOUT) {
    int n = i / FIN, k = i % FIN;
    Wt[i] = (__bf16)W[k * FOUT + n];
  }
}

// Wt[n][k] over K=2*CH: first CH rows from Wroot, next CH rows from Wrel.
template <int CH, int COUT>
__global__ void k_wcat(const float* __restrict__ Wroot, const float* __restrict__ Wrel,
                       __bf16* __restrict__ Wt) {
  int i = blockIdx.x * blockDim.x + threadIdx.x;
  if (i < COUT * 2 * CH) {
    int n = i / (2 * CH), k = i % (2 * CH);
    float v = (k < CH) ? Wroot[k * COUT + n] : Wrel[(k - CH) * COUT + n];
    Wt[i] = (__bf16)v;
  }
}

// per-channel finalize: scale = g*rsqrt(var+eps), shift = be - mean*scale
template <int NCH>
__global__ void k_bnfinal(const float* __restrict__ sum, const float* __restrict__ sq,
                          const float* __restrict__ g, const float* __restrict__ be,
                          float inv, float* __restrict__ sc, float* __restrict__ sh) {
  int t = threadIdx.x;
  if (t < NCH) {
    float m = sum[t] * inv;
    float v = sq[t] * inv - m * m;
    float s = g[t] * rsqrtf(v + BN_EPS);
    sc[t] = s;
    sh[t] = be[t] - m * s;
  }
}

// per-electrode (row % 19) sums over [NROWS][C] bf16; one row per thread.
template <int C>
__global__ void k_rowstats(const __bf16* __restrict__ H, float* __restrict__ sum,
                           float* __restrict__ sq) {
  __shared__ float ls[NEL], lq[NEL];
  int t = threadIdx.x;
  if (t < NEL) { ls[t] = 0.f; lq[t] = 0.f; }
  __syncthreads();
  long r = (long)blockIdx.x * blockDim.x + t;  // grid covers exactly NROWS
  const __bf16* row = H + r * C;
  float s = 0.f, q = 0.f;
#pragma unroll 8
  for (int c = 0; c < C; ++c) {
    float v = (float)row[c];
    s += v; q += v * v;
  }
  int e = (int)(r % NEL);
  atomicAdd(&ls[e], s);
  atomicAdd(&lq[e], q);
  __syncthreads();
  if (t < NEL) {
    atomicAdd(&sum[t], ls[t]);
    atomicAdd(&sq[t], lq[t]);
  }
}

// per-column sums over [NROWS][32] bf16
__global__ void k_colstats32(const __bf16* __restrict__ H, float* __restrict__ sum,
                             float* __restrict__ sq) {
  __shared__ float ls[32], lq[32];
  int t = threadIdx.x;
  if (t < 32) { ls[t] = 0.f; lq[t] = 0.f; }
  __syncthreads();
  int c = t & 31;
  int sub = t >> 5;  // 8 row-groups per block
  float s = 0.f, q = 0.f;
  for (long r = (long)blockIdx.x * 8 + sub; r < NROWS; r += (long)gridDim.x * 8) {
    float v = (float)H[r * 32 + c];
    s += v; q += v * v;
  }
  atomicAdd(&ls[c], s);
  atomicAdd(&lq[c], q);
  __syncthreads();
  if (t < 32) { atomicAdd(&sum[t], ls[t]); atomicAdd(&sq[t], lq[t]); }
}

// ------------------------- FC1: pool(5) + GEMM 800->256 --------------------
// block: 256 threads (8 waves), 32 rows, all 256 output cols, K streamed.
__global__ void k_fc1(const float* __restrict__ x, const __bf16* __restrict__ W2t,
                      const float* __restrict__ bias, __bf16* __restrict__ Out) {
  constexpr int CIN = 800, COUT = 256, MT = 32, NT = COUT / 16, KC = 64;
  extern __shared__ char smem[];
  __bf16* As = (__bf16*)smem;                  // [32][800]
  __bf16* Ws = (__bf16*)(smem + MT * CIN * 2); // [256][64] chunk
  const unsigned ws_lds = (unsigned)(unsigned long long)(uintptr_t)Ws;
  const int t = threadIdx.x;
  const long row0 = (long)blockIdx.x * MT;

  // stage A: pool groups of 5 raw fp32 -> bf16 (compute path, via VGPRs)
  for (int p = t; p < MT * CIN; p += 256) {
    int r = p / CIN, k = p % CIN;
    const float* s = x + (row0 + r) * 4000L + (long)k * 5;
    float v = (s[0] + s[1] + s[2] + s[3] + s[4]) * 0.2f;
    As[r * CIN + k] = (__bf16)v;
  }

  const int wave = t >> 5, lane = t & 31;
  const int lrow = lane & 15, lhi = lane >> 4;
  v8f acc[4] = {};

  for (int k0 = 0; k0 < CIN; k0 += KC) {
    __syncthreads();  // previous chunk fully consumed before overwrite
    // stage W chunk [256][64] bf16 via async global->LDS (ASYNCcnt)
    {
      int n = t;  // one output row per thread
      const unsigned long long gbase =
          (unsigned long long)(uintptr_t)(W2t + (long)n * CIN + k0);
      const unsigned lbase = ws_lds + (unsigned)(n * KC) * 2u;
#pragma unroll
      for (int j = 0; j < KC / 8; ++j)
        async_g2l_b128(lbase + (unsigned)j * 16u, gbase + (unsigned long long)j * 16ull);
      if (k0 + KC < CIN)
        __builtin_prefetch(W2t + (long)n * CIN + k0 + KC, 0, 1);
    }
    wait_async0();
    __syncthreads();
#pragma unroll
    for (int kk = 0; kk < KC; kk += 32) {
#pragma unroll
      for (int i = 0; i < 4; ++i) {
        int tile = wave * 4 + i;
        int rt = tile / NT, ct = tile % NT;
        ABf a, b;
        const __bf16* ap = &As[(rt * 16 + lrow) * CIN + k0 + kk + lhi * 8];
        a.h[0] = *(const v8bf*)ap;
        a.h[1] = *(const v8bf*)(ap + 16);
        const __bf16* bp = &Ws[(ct * 16 + lrow) * KC + kk + lhi * 8];
        b.h[0] = *(const v8bf*)bp;
        b.h[1] = *(const v8bf*)(bp + 16);
        acc[i] = __builtin_amdgcn_wmma_f32_16x16x32_bf16(false, a.v, false, b.v,
                                                         (short)0, acc[i], false, false);
      }
    }
  }
  // epilogue: bias + ReLU -> bf16
#pragma unroll
  for (int i = 0; i < 4; ++i) {
    int tile = wave * 4 + i;
    int rt = tile / NT, ct = tile % NT;
    int n = ct * 16 + lrow;
    float bn = bias[n];
#pragma unroll
    for (int j = 0; j < 8; ++j) {
      long gr = row0 + rt * 16 + lhi * 8 + j;
      float v = acc[i][j] + bn;
      Out[gr * COUT + n] = (__bf16)(v > 0.f ? v : 0.f);
    }
  }
}

// ------------------- generic WMMA GEMM: out = relu(A@Wt^T + b) -------------
// A: [R][CIN] bf16 (optional per-row-electrode affine), Wt: [COUT][CIN] bf16.
template <int CIN, int COUT, bool ROW_AFFINE>
__global__ void k_fc(const __bf16* __restrict__ A, const __bf16* __restrict__ Wt,
                     const float* __restrict__ bias, const float* __restrict__ sc,
                     const float* __restrict__ sh, __bf16* __restrict__ Out) {
  constexpr int MT = 64, RT = MT / 16, NT = COUT / 16, WAVES = 8;
  constexpr int TPW = (RT * NT) / WAVES;
  extern __shared__ char smem[];
  __bf16* As = (__bf16*)smem;                   // [MT][CIN]
  __bf16* Ws = (__bf16*)(smem + MT * CIN * 2);  // [COUT][CIN]
  const unsigned ws_lds = (unsigned)(unsigned long long)(uintptr_t)Ws;
  const int t = threadIdx.x;
  const long row0 = (long)blockIdx.x * MT;

  // stage full W via async global->LDS copies (pure bf16 copy, no VGPR data)
  for (int p = t; p < COUT * CIN / 8; p += 256)
    async_g2l_b128(ws_lds + (unsigned)p * 16u,
                   (unsigned long long)(uintptr_t)Wt + (unsigned long long)p * 16ull);

  // stage A with optional per-row(e) BN affine (compute path)
  for (int p = t; p < MT * CIN; p += 256) {
    int r = p / CIN, c = p % CIN;
    long gr = row0 + r;
    float v = (float)A[gr * CIN + c];
    if constexpr (ROW_AFFINE) {
      int e = (int)(gr % NEL);
      v = v * sc[e] + sh[e];
    }
    As[p] = (__bf16)v;
  }
  wait_async0();
  __syncthreads();

  const int wave = t >> 5, lane = t & 31;
  const int lrow = lane & 15, lhi = lane >> 4;
  v8f acc[TPW] = {};
  for (int k0 = 0; k0 < CIN; k0 += 32) {
#pragma unroll
    for (int i = 0; i < TPW; ++i) {
      int tile = wave * TPW + i;
      int rt = tile / NT, ct = tile % NT;
      ABf a, b;
      const __bf16* ap = &As[(rt * 16 + lrow) * CIN + k0 + lhi * 8];
      a.h[0] = *(const v8bf*)ap;
      a.h[1] = *(const v8bf*)(ap + 16);
      const __bf16* bp = &Ws[(ct * 16 + lrow) * CIN + k0 + lhi * 8];
      b.h[0] = *(const v8bf*)bp;
      b.h[1] = *(const v8bf*)(bp + 16);
      acc[i] = __builtin_amdgcn_wmma_f32_16x16x32_bf16(false, a.v, false, b.v,
                                                       (short)0, acc[i], false, false);
    }
  }
#pragma unroll
  for (int i = 0; i < TPW; ++i) {
    int tile = wave * TPW + i;
    int rt = tile / NT, ct = tile % NT;
    int n = ct * 16 + lrow;
    float bn = bias[n];
#pragma unroll
    for (int j = 0; j < 8; ++j) {
      long gr = row0 + rt * 16 + lhi * 8 + j;
      float v = acc[i][j] + bn;
      Out[gr * COUT + n] = (__bf16)(v > 0.f ? v : 0.f);
    }
  }
}

// ---------------- graph scatter: A = [bn(h) | segsum(ew*bn(h))] ------------
// conv1: C=64, per-row(e) affine. One block (64 threads) per graph.
__global__ void k_gather1(const __bf16* __restrict__ H, const float* __restrict__ sc,
                          const float* __restrict__ sh, const long long* __restrict__ ei,
                          const float* __restrict__ ew, __bf16* __restrict__ Acat) {
  constexpr int C = 64;
  __shared__ float hn[NEL * C];
  __shared__ float agg[NEL * C];
  const int f = threadIdx.x;
  const int g = blockIdx.x;
  const long rbase = (long)g * NEL;
#pragma unroll
  for (int e = 0; e < NEL; ++e) {
    float v = (float)H[(rbase + e) * C + f] * sc[e] + sh[e];
    hn[e * C + f] = v;
    agg[e * C + f] = 0.f;
  }
  __syncthreads();
  const long long* esrc = ei;
  const long long* edst = ei + (long)NB * EPG;
#pragma unroll 4
  for (int i = 0; i < EPG; ++i) {
    int s = (int)esrc[(long)g * EPG + i] - g * NEL;
    int d = (int)edst[(long)g * EPG + i] - g * NEL;
    agg[d * C + f] += ew[i] * hn[s * C + f];  // thread owns column f: no races
  }
#pragma unroll
  for (int e = 0; e < NEL; ++e) {
    Acat[(rbase + e) * (2 * C) + f] = (__bf16)hn[e * C + f];
    Acat[(rbase + e) * (2 * C) + C + f] = (__bf16)agg[e * C + f];
  }
}

// conv2: C=32, per-COLUMN affine. One block (32 threads) per graph.
__global__ void k_gather2(const __bf16* __restrict__ H, const float* __restrict__ sc,
                          const float* __restrict__ sh, const long long* __restrict__ ei,
                          const float* __restrict__ ew, __bf16* __restrict__ Acat) {
  constexpr int C = 32;
  __shared__ float hn[NEL * C];
  __shared__ float agg[NEL * C];
  const int f = threadIdx.x;
  const int g = blockIdx.x;
  const long rbase = (long)g * NEL;
  const float s0 = sc[f], h0 = sh[f];
#pragma unroll
  for (int e = 0; e < NEL; ++e) {
    float v = (float)H[(rbase + e) * C + f] * s0 + h0;
    hn[e * C + f] = v;
    agg[e * C + f] = 0.f;
  }
  __syncthreads();
  const long long* esrc = ei;
  const long long* edst = ei + (long)NB * EPG;
#pragma unroll 4
  for (int i = 0; i < EPG; ++i) {
    int s = (int)esrc[(long)g * EPG + i] - g * NEL;
    int d = (int)edst[(long)g * EPG + i] - g * NEL;
    agg[d * C + f] += ew[i] * hn[s * C + f];
  }
#pragma unroll
  for (int e = 0; e < NEL; ++e) {
    Acat[(rbase + e) * (2 * C) + f] = (__bf16)hn[e * C + f];
    Acat[(rbase + e) * (2 * C) + C + f] = (__bf16)agg[e * C + f];
  }
}

// -------- final: bn -> segment_max(19) -> FC 32x32 + relu -> 3 heads -------
__global__ void k_head(const __bf16* __restrict__ H, const float* __restrict__ sc,
                       const float* __restrict__ sh, const float* __restrict__ W5,
                       const float* __restrict__ b5, const float* __restrict__ Wm,
                       const float* __restrict__ bm, const float* __restrict__ Wb,
                       const float* __restrict__ bbias, const float* __restrict__ Wd,
                       const float* __restrict__ bd, float* __restrict__ out) {
  __shared__ float pooled[32];
  __shared__ float z[32];
  const int t = threadIdx.x;  // 32 threads
  const int g = blockIdx.x;
  const float s0 = sc[t], h0 = sh[t];
  float m = -3.4e38f;
  const long base = (long)g * NEL * 32;
#pragma unroll
  for (int e = 0; e < NEL; ++e) {
    float v = (float)H[base + e * 32 + t] * s0 + h0;
    m = fmaxf(m, v);
  }
  pooled[t] = m;
  __syncthreads();
  float a = b5[t];
#pragma unroll
  for (int k = 0; k < 32; ++k) a += pooled[k] * W5[k * 32 + t];
  z[t] = fmaxf(a, 0.f);
  __syncthreads();
  if (t < 3) {
    float o = bm[t];
#pragma unroll
    for (int k = 0; k < 32; ++k) o += z[k] * Wm[k * 3 + t];
    out[(long)g * 3 + t] = o;
  } else if (t < 5) {
    int j = t - 3;
    float o = bbias[j];
#pragma unroll
    for (int k = 0; k < 32; ++k) o += z[k] * Wb[k * 2 + j];
    out[(long)NB * 3 + (long)g * 2 + j] = o;
  } else if (t < 7) {
    int j = t - 5;
    float o = bd[j];
#pragma unroll
    for (int k = 0; k < 32; ++k) o += z[k] * Wd[k * 2 + j];
    out[(long)NB * 5 + (long)g * 2 + j] = o;
  }
}

// --------------------------------- launch ----------------------------------
extern "C" void kernel_launch(void* const* d_in, const int* in_sizes, int n_in,
                              void* d_out, int out_size, void* d_ws, size_t ws_size,
                              hipStream_t stream) {
  const float* x      = (const float*)d_in[0];
  const float* W2     = (const float*)d_in[1];
  const float* b2     = (const float*)d_in[2];
  const float* W3     = (const float*)d_in[3];
  const float* b3     = (const float*)d_in[4];
  const float* W4     = (const float*)d_in[5];
  const float* b4     = (const float*)d_in[6];
  const float* g3     = (const float*)d_in[7];
  const float* be3    = (const float*)d_in[8];
  const float* g4     = (const float*)d_in[9];
  const float* be4    = (const float*)d_in[10];
  const float* g5     = (const float*)d_in[11];
  const float* be5    = (const float*)d_in[12];
  const float* Wr1    = (const float*)d_in[13];
  const float* br1    = (const float*)d_in[14];
  const float* Wroot1 = (const float*)d_in[15];
  const float* ew1    = (const float*)d_in[16];
  const float* Wr2    = (const float*)d_in[17];
  const float* br2    = (const float*)d_in[18];
  const float* Wroot2 = (const float*)d_in[19];
  const float* ew2    = (const float*)d_in[20];
  const float* g6     = (const float*)d_in[21];
  const float* be6    = (const float*)d_in[22];
  const float* g7     = (const float*)d_in[23];
  const float* be7    = (const float*)d_in[24];
  const float* W5     = (const float*)d_in[25];
  const float* b5     = (const float*)d_in[26];
  const float* Wm     = (const float*)d_in[27];
  const float* bm     = (const float*)d_in[28];
  const float* Wb     = (const float*)d_in[29];
  const float* bbv    = (const float*)d_in[30];
  const float* Wd     = (const float*)d_in[31];
  const float* bd     = (const float*)d_in[32];
  const long long* ei = (const long long*)d_in[33];
  float* out = (float*)d_out;

  char* ws = (char*)d_ws;
  float* st = (float*)ws;
  float *s3_sum = st +   0, *s3_sq = st +  32, *s3_sc = st +  64, *s3_sh = st +  96;
  float *s4_sum = st + 128, *s4_sq = st + 160, *s4_sc = st + 192, *s4_sh = st + 224;
  float *s5_sum = st + 256, *s5_sq = st + 288, *s5_sc = st + 320, *s5_sh = st + 352;
  float *c6_sum = st + 384, *c6_sq = st + 416, *c6_sc = st + 448, *c6_sh = st + 480;
  float *c7_sum = st + 512, *c7_sq = st + 544, *c7_sc = st + 576, *c7_sh = st + 608;
  __bf16* w2t = (__bf16*)(ws + OFF_W2T);
  __bf16* w3t = (__bf16*)(ws + OFF_W3T);
  __bf16* w4t = (__bf16*)(ws + OFF_W4T);
  __bf16* wc1 = (__bf16*)(ws + OFF_WC1);
  __bf16* wc2 = (__bf16*)(ws + OFF_WC2);
  __bf16* h1 = (__bf16*)(ws + OFF_H1);
  __bf16* h2 = (__bf16*)(ws + OFF_H2);
  __bf16* h3 = (__bf16*)(ws + OFF_H3);
  __bf16* a4 = (__bf16*)(ws + OFF_A4);
  __bf16* h4 = (__bf16*)(ws + OFF_H4);
  __bf16* a5 = (__bf16*)(ws + OFF_A5);
  __bf16* h5 = (__bf16*)(ws + OFF_H5);

  // 0) zero stats + weight prep
  k_zero<<<3, 256, 0, stream>>>(st, 640);
  k_wt<800, 256><<<(256 * 800 + 255) / 256, 256, 0, stream>>>(W2, w2t);
  k_wt<256, 128><<<(128 * 256 + 255) / 256, 256, 0, stream>>>(W3, w3t);
  k_wt<128, 64><<<(64 * 128 + 255) / 256, 256, 0, stream>>>(W4, w4t);
  k_wcat<64, 32><<<(32 * 128 + 255) / 256, 256, 0, stream>>>(Wroot1, Wr1, wc1);
  k_wcat<32, 32><<<(32 * 64 + 255) / 256, 256, 0, stream>>>(Wroot2, Wr2, wc2);

  // 1) pool + FC1 (800->256) + ReLU
  k_fc1<<<NROWS / 32, 256, 32 * 800 * 2 + 256 * 64 * 2, stream>>>(x, w2t, b2, h1);
  k_rowstats<256><<<NROWS / 256, 256, 0, stream>>>(h1, s3_sum, s3_sq);
  k_bnfinal<NEL><<<1, 32, 0, stream>>>(s3_sum, s3_sq, g3, be3, 1.f / (4096.f * 256.f), s3_sc, s3_sh);

  // 2) FC2 (256->128)
  k_fc<256, 128, true><<<NROWS / 64, 256, 64 * 256 * 2 + 128 * 256 * 2, stream>>>(
      h1, w3t, b3, s3_sc, s3_sh, h2);
  k_rowstats<128><<<NROWS / 256, 256, 0, stream>>>(h2, s4_sum, s4_sq);
  k_bnfinal<NEL><<<1, 32, 0, stream>>>(s4_sum, s4_sq, g4, be4, 1.f / (4096.f * 128.f), s4_sc, s4_sh);

  // 3) FC3 (128->64)
  k_fc<128, 64, true><<<NROWS / 64, 256, 64 * 128 * 2 + 64 * 128 * 2, stream>>>(
      h2, w4t, b4, s4_sc, s4_sh, h3);
  k_rowstats<64><<<NROWS / 256, 256, 0, stream>>>(h3, s5_sum, s5_sq);
  k_bnfinal<NEL><<<1, 32, 0, stream>>>(s5_sum, s5_sq, g5, be5, 1.f / (4096.f * 64.f), s5_sc, s5_sh);

  // 4) GraphConv1: gather + GEMM([hn|agg] @ [Wroot1;Wr1]) + ReLU
  k_gather1<<<NB, 64, 0, stream>>>(h3, s5_sc, s5_sh, ei, ew1, a4);
  k_fc<128, 32, false><<<NROWS / 64, 256, 64 * 128 * 2 + 32 * 128 * 2, stream>>>(
      a4, wc1, br1, nullptr, nullptr, h4);
  k_colstats32<<<NROWS / 256, 256, 0, stream>>>(h4, c6_sum, c6_sq);
  k_bnfinal<32><<<1, 32, 0, stream>>>(c6_sum, c6_sq, g6, be6, 1.f / (float)NROWS, c6_sc, c6_sh);

  // 5) GraphConv2
  k_gather2<<<NB, 32, 0, stream>>>(h4, c6_sc, c6_sh, ei, ew2, a5);
  k_fc<64, 32, false><<<NROWS / 64, 256, 64 * 64 * 2 + 32 * 64 * 2, stream>>>(
      a5, wc2, br2, nullptr, nullptr, h5);
  k_colstats32<<<NROWS / 256, 256, 0, stream>>>(h5, c7_sum, c7_sq);
  k_bnfinal<32><<<1, 32, 0, stream>>>(c7_sum, c7_sq, g7, be7, 1.f / (float)NROWS, c7_sc, c7_sh);

  // 6) max-pool over electrodes + FC 32x32 + heads
  k_head<<<NB, 32, 0, stream>>>(h5, c7_sc, c7_sh, W5, b5, Wm, bm, Wb, bbv, Wd, bd, out);
}
